// clacClass_57483842290145
// MI455X (gfx1250) — matrix operand order repaired and checked by reading, not compiled, not run on hardware
//
#include <hip/hip_runtime.h>

typedef __attribute__((ext_vector_type(16))) _Float16 v16h;
typedef __attribute__((ext_vector_type(8)))  float    v8f;

#define N_SIG   4096
#define M_ROWS  2112         // 33 tiles of 64 rows, covers the 2049 needed outputs
#define HALF_N  2049
#define KT      64
#define NB      32           // N blocks of 128 columns
#define LDSS    72           // LDS row stride in halfs (16B aligned, bank-friendly)
#define PI_APPROX 3.14f
#define FPS 30.0f

// Low 32 bits of a flat pointer to LDS == LDS byte offset (AS3 -> flat cast
// places the LDS address in addr[31:0]; async VDST wants exactly that).
static __device__ __forceinline__ uint32_t lds_addr_of(const void* p) {
    return (uint32_t)(uintptr_t)p;
}

// ---------------- Kernel 1: per-row means ----------------
__global__ void row_mean_kernel(const float* __restrict__ x, float* __restrict__ means) {
    const int row = blockIdx.x;
    const float* xr = x + (size_t)row * N_SIG;
    __shared__ float red[256];
    float s = 0.0f;
    for (int c = threadIdx.x; c < N_SIG; c += 256) s += xr[c];
    red[threadIdx.x] = s;
    __syncthreads();
    for (int st = 128; st > 0; st >>= 1) {
        if (threadIdx.x < st) red[threadIdx.x] += red[threadIdx.x + st];
        __syncthreads();
    }
    if (threadIdx.x == 0) means[row] = red[0] * (1.0f / N_SIG);
}

// ------- Kernel 2: center + convert f16 + transpose (64x64 LDS tiles) -------
// XhT[n][k] = (f16)(x[k][n] - mean[k]); coalesced reads and writes.
__global__ void __launch_bounds__(256)
center_transpose_kernel(const float* __restrict__ x, const float* __restrict__ means,
                        _Float16* __restrict__ XhT) {
    __shared__ __align__(16) _Float16 t[64][LDSS];
    const int k0 = blockIdx.x * 64;
    const int n0 = blockIdx.y * 64;
    const int tid = threadIdx.x;
    #pragma unroll
    for (int it = 0; it < 16; ++it) {
        const int idx = it * 256 + tid;
        const int r = idx >> 6, c = idx & 63;          // r: k-local, c: n-local
        t[r][c] = (_Float16)(x[(size_t)(k0 + r) * N_SIG + n0 + c] - means[k0 + r]);
    }
    __syncthreads();
    #pragma unroll
    for (int it = 0; it < 16; ++it) {
        const int idx = it * 256 + tid;
        const int rr = idx >> 6, cc = idx & 63;        // rr: n-local, cc: k-local
        XhT[(size_t)(n0 + rr) * N_SIG + k0 + cc] = t[cc][rr];
    }
}

// ---------------- Kernel 3: cos/sin DFT tables in f16 ----------------
__global__ void trig_f16_kernel(_Float16* __restrict__ Ch, _Float16* __restrict__ Sh) {
    const int t = blockIdx.x * 256 + threadIdx.x;
    if (t >= M_ROWS * N_SIG) return;
    const int i = t >> 12;
    const int j = t & (N_SIG - 1);
    // i*j <= 2111*4095 < 2^24 -> exact in f32
    const float theta = -(2.0f * PI_APPROX / (float)N_SIG) * (float)(i * j);
    Ch[t] = (_Float16)cosf(theta);
    Sh[t] = (_Float16)sinf(theta);
}

// ------ Kernel 4: fused WMMA GEMM (R & I) + per-row partial argmax ------
// 256 threads = 8 waves. Block tile 64(M) x 128(N), KT=64, double-buffered
// async-to-LDS staging (8 x 512B per wave per stage), statically unrolled x2.
// Wave (wm = w&3, wn = w>>2) computes rows wm*16..+15, cols wn*64..+63
// -> 4 N-subtiles x {R,I} accumulators.
__global__ void __launch_bounds__(256)
dft_argmax_kernel(const _Float16* __restrict__ Ch, const _Float16* __restrict__ Sh,
                  const _Float16* __restrict__ XhT,
                  float* __restrict__ pVal, int* __restrict__ pIdx) {
    __shared__ __align__(16) _Float16 Ac[2][64 * LDSS];
    __shared__ __align__(16) _Float16 As[2][64 * LDSS];
    __shared__ __align__(16) _Float16 Bs[2][128 * LDSS];
    __shared__ float lBV[2][64];
    __shared__ int   lBI[2][64];

    const int tid  = threadIdx.x;
    const int lane = tid & 31;
    const int wave = tid >> 5;
    const int wm = wave & 3;
    const int wn = wave >> 2;
    const int mb = blockIdx.x;   // 0..32
    const int nb = blockIdx.y;   // 0..31

    v8f accR[4] = {{}, {}, {}, {}};
    v8f accI[4] = {{}, {}, {}, {}};

    const _Float16* gA0 = Ch  + (size_t)(mb * 64) * N_SIG;
    const _Float16* gA1 = Sh  + (size_t)(mb * 64) * N_SIG;
    const _Float16* gB  = XhT + (size_t)(nb * 128) * N_SIG;
    const uint32_t ldsAc[2] = { lds_addr_of(&Ac[0][0]), lds_addr_of(&Ac[1][0]) };
    const uint32_t ldsAs[2] = { lds_addr_of(&As[0][0]), lds_addr_of(&As[1][0]) };
    const uint32_t ldsBs[2] = { lds_addr_of(&Bs[0][0]), lds_addr_of(&Bs[1][0]) };

    // Stage issue: A (2 x 64x64 halfs) + B (128x64 halfs) = 32KB -> 64 wave-wide
    // 512B transfers -> 8 global_load_async_to_lds_b128 per wave.
    auto issue = [&](int kt, uint32_t lAc, uint32_t lAs, uint32_t lBsb) {
        #pragma unroll
        for (int q = 0; q < 8; ++q) {
            const int id = wave * 8 + q;      // 0..63 (wave-uniform)
            const int c16 = lane & 7;
            const _Float16* gbase;
            uint32_t lbase;
            int r;
            if (id < 16)      { gbase = gA0; lbase = lAc;  r = id * 4 + (lane >> 3); }
            else if (id < 32) { gbase = gA1; lbase = lAs;  r = (id - 16) * 4 + (lane >> 3); }
            else              { gbase = gB;  lbase = lBsb; r = (id - 32) * 4 + (lane >> 3); }
            const unsigned long long gaddr =
                (unsigned long long)(uintptr_t)(gbase + (size_t)r * N_SIG + kt + c16 * 8);
            const uint32_t laddr = lbase + (uint32_t)(r * LDSS + c16 * 8) * 2u;
            asm volatile("global_load_async_to_lds_b128 %0, %1, off"
                         :: "v"(laddr), "v"(gaddr) : "memory");
        }
    };

    auto compute = [&](const _Float16* pAc, const _Float16* pAs, const _Float16* pBs) {
        #pragma unroll
        for (int ks = 0; ks < KT; ks += 32) {
            // A fragment (16x32 f16): lane L -> M=L%16; K(e)=(e/8)*16+(L>=16?8:0)+e%8
            const int off  = (lane >= 16) ? 8 : 0;
            const int arow = (wm * 16 + (lane & 15)) * LDSS;
            union { v16h v; uint4 u[2]; } fc, fs;
            fc.u[0] = *(const uint4*)(pAc + arow + ks + off);
            fc.u[1] = *(const uint4*)(pAc + arow + ks + 16 + off);
            fs.u[0] = *(const uint4*)(pAs + arow + ks + off);
            fs.u[1] = *(const uint4*)(pAs + arow + ks + 16 + off);
            // B fragment (32x16 f16): lane L -> N=L%16; K(e)=(e/8)*16+(L>=16?8:0)+e%8
            const int klo = (lane >= 16) ? 8 : 0;
            #pragma unroll
            for (int nt = 0; nt < 4; ++nt) {
                const int brow = (wn * 64 + nt * 16 + (lane & 15)) * LDSS;
                union { v16h v; uint4 u[2]; } fb;
                fb.u[0] = *(const uint4*)(pBs + brow + ks + klo);
                fb.u[1] = *(const uint4*)(pBs + brow + ks + klo + 16);
                accR[nt] = __builtin_amdgcn_wmma_f32_16x16x32_f16(
                    false, fc.v, false, fb.v, (short)0, accR[nt], false, false);
                accI[nt] = __builtin_amdgcn_wmma_f32_16x16x32_f16(
                    false, fs.v, false, fb.v, (short)0, accI[nt], false, false);
            }
        }
    };

    // Prologue: stage tile 0 into buffer 0.
    issue(0, ldsAc[0], ldsAs[0], ldsBs[0]);

    for (int kt = 0; kt < N_SIG; kt += 2 * KT) {
        // ---- stage A: prefetch kt+KT into buf1 (always valid), compute buf0 ----
        issue(kt + KT, ldsAc[1], ldsAs[1], ldsBs[1]);
        asm volatile("s_wait_asynccnt 0x8" ::: "memory");   // buf0's 8 done (in-order)
        __syncthreads();
        compute(Ac[0], As[0], Bs[0]);
        __syncthreads();
        // ---- stage B: prefetch kt+2*KT into buf0 (if any), compute buf1 ----
        if (kt + 2 * KT < N_SIG) {
            issue(kt + 2 * KT, ldsAc[0], ldsAs[0], ldsBs[0]);
            asm volatile("s_wait_asynccnt 0x8" ::: "memory");
        } else {
            asm volatile("s_wait_asynccnt 0x0" ::: "memory");
        }
        __syncthreads();
        compute(Ac[1], As[1], Bs[1]);
        __syncthreads();
    }

    // Epilogue: amplitude^2 = r^2 + i^2 (monotone -> same argmax as reference).
    // C/D layout: acc[v] holds M = v + 8*(lane/16), N = lane%16 within the tile.
    #pragma unroll
    for (int v = 0; v < 8; ++v) {
        float bV = -1.0f; int bI = 0x7fffffff;
        #pragma unroll
        for (int nt = 0; nt < 4; ++nt) {
            const float r  = accR[nt][v];
            const float im = accI[nt][v];
            float a2 = r * r + im * im;
            int idx = nb * 128 + wn * 64 + nt * 16 + (lane & 15);
            #pragma unroll
            for (int m = 8; m >= 1; m >>= 1) {       // butterfly within 16-lane halves
                const float ov = __shfl_xor(a2, m, 32);
                const int   oi = __shfl_xor(idx, m, 32);
                if (ov > a2 || (ov == a2 && oi < idx)) { a2 = ov; idx = oi; }
            }
            if (a2 > bV || (a2 == bV && idx < bI)) { bV = a2; bI = idx; }
        }
        if ((lane & 15) == 0) {
            const int ml = wm * 16 + v + 8 * (lane >> 4);
            lBV[wn][ml] = bV; lBI[wn][ml] = bI;
        }
    }
    __syncthreads();
    if (tid < 64) {
        float v0 = lBV[0][tid]; int i0 = lBI[0][tid];
        const float v1 = lBV[1][tid]; const int i1 = lBI[1][tid];
        if (v1 > v0 || (v1 == v0 && i1 < i0)) { v0 = v1; i0 = i1; }
        const int m = mb * 64 + tid;
        pVal[(size_t)nb * M_ROWS + m] = v0;
        pIdx[(size_t)nb * M_ROWS + m] = i0;
    }
}

// ---------------- Kernel 5: final argmax reduce over NB N-blocks ----------------
__global__ void final_reduce_kernel(const float* __restrict__ pVal,
                                    const int* __restrict__ pIdx,
                                    float* __restrict__ out) {
    const int m = blockIdx.x * 256 + threadIdx.x;
    if (m >= HALF_N) return;
    float bV = -1.0f; int bI = 0x7fffffff;
    for (int nbk = 0; nbk < NB; ++nbk) {
        const float v = pVal[(size_t)nbk * M_ROWS + m];
        const int   i = pIdx[(size_t)nbk * M_ROWS + m];
        if (v > bV || (v == bV && i < bI)) { bV = v; bI = i; }
    }
    // hr = freq[idx] * 60, freq[j] = j * FPS / Ns
    out[m] = (float)bI * (FPS / (float)N_SIG) * 60.0f;
}

// ---------------- Launch ----------------
extern "C" void kernel_launch(void* const* d_in, const int* in_sizes, int n_in,
                              void* d_out, int out_size, void* d_ws, size_t ws_size,
                              hipStream_t stream) {
    const float* x = (const float*)d_in[0];
    float* out = (float*)d_out;
    char* ws = (char*)d_ws;

    // Workspace layout (bytes), all 256B-aligned:
    _Float16* XhT  = (_Float16*)(ws);                 // 4096*4096*2 = 33,554,432
    _Float16* Ch   = (_Float16*)(ws + 33554432);      // 2112*4096*2 = 17,301,504
    _Float16* Sh   = (_Float16*)(ws + 50855936);      // 17,301,504
    float*    means= (float*)   (ws + 68157440);      //      16,384
    float*    pVal = (float*)   (ws + 68173824);      // 32*2112*4 = 270,336
    int*      pIdx = (int*)     (ws + 68444160);      //     270,336
    // total ~68.7 MB

    row_mean_kernel<<<N_SIG, 256, 0, stream>>>(x, means);
    dim3 tg(N_SIG / 64, N_SIG / 64);                  // 64 x 64
    center_transpose_kernel<<<tg, 256, 0, stream>>>(x, means, XhT);
    trig_f16_kernel<<<(M_ROWS * N_SIG + 255) / 256, 256, 0, stream>>>(Ch, Sh);
    dim3 grid(M_ROWS / 64, N_SIG / 128);              // 33 x 32
    dft_argmax_kernel<<<grid, 256, 0, stream>>>(Ch, Sh, XhT, pVal, pIdx);
    final_reduce_kernel<<<(HALF_N + 255) / 256, 256, 0, stream>>>(pVal, pIdx, out);
}
